// TieredLayerMemory_32744830665512
// MI455X (gfx1250) — matrix operand order repaired and checked by reading, not compiled, not run on hardware
//
#include <hip/hip_runtime.h>
#include <hip/hip_bf16.h>
#include <math.h>

#define DIM     1024
#define S_SIZE  512
#define M_SIZE  4096
#define L_SIZE  8192
#define NCAND   16384
#define KPROM   128

typedef __attribute__((ext_vector_type(16))) __bf16 v16bf;
typedef __attribute__((ext_vector_type(8)))  float  v8f;

// ---- workspace byte offsets (all 16B aligned where needed) ----
#define WS_SLOTS  0u        // 8 x u64 packed (val,idx) atomics
#define WS_SCAL   64u       // f32 scalars: [0]=mean(mu) [1]=mean(lu) [2]=||cand||
#define WS_ISCAL  96u       // i32 scalars: [0]=best_s
#define WS_NORMS  128u      // 16384 f32 candidate norms
#define WS_TOPIDX 65664u    // 128 i32
#define WS_LMEAN  66176u    // 1024 f32
#define WS_Q      70272u    // 1024 f32
#define WS_T      74368u    // 1024 f32
#define WS_CN     78464u    // 1024 f32 normalized candidate
#define WS_MRN    82560u    // 4096 f32 row norms of m_memory
#define WS_MBF    98944u    // 4096*1024 bf16 normalized M rows (8 MB)

__device__ __forceinline__ unsigned long long packMax(float v, unsigned idx) {
  unsigned u = __float_as_uint(v);
  u = (u & 0x80000000u) ? ~u : (u | 0x80000000u);   // order-preserving float->uint
  return ((unsigned long long)u << 32) | (unsigned long long)(0xFFFFFFFFu - idx);
}
__device__ __forceinline__ float unpackVal(unsigned long long k) {
  unsigned u = (unsigned)(k >> 32);
  u = (u & 0x80000000u) ? (u ^ 0x80000000u) : ~u;
  return __uint_as_float(u);
}
__device__ __forceinline__ unsigned unpackIdx(unsigned long long k) {
  return 0xFFFFFFFFu - (unsigned)(k & 0xFFFFFFFFu);
}

// ---------------------------------------------------------------- small utils
__global__ void zero_slots_k(unsigned long long* slots) {
  if (threadIdx.x < 8) slots[threadIdx.x] = 0ull;
}

__global__ void copy4_k(float4* __restrict__ dst, const float4* __restrict__ src, size_t n4) {
  size_t i = (size_t)blockIdx.x * blockDim.x + threadIdx.x;
  size_t stride = (size_t)gridDim.x * blockDim.x;
  for (; i < n4; i += stride) dst[i] = src[i];
}

// ---------------------------------------------------------------- S tier
__global__ void cand_norms_k(const float* __restrict__ cand, float* __restrict__ norms) {
  __shared__ float red[256];
  int r = blockIdx.x;
  const float* row = cand + (size_t)r * DIM;
  float s = 0.f;
  for (int c = threadIdx.x; c < DIM; c += 256) { float x = row[c]; s += x * x; }
  red[threadIdx.x] = s; __syncthreads();
  for (int o = 128; o > 0; o >>= 1) {
    if (threadIdx.x < o) red[threadIdx.x] += red[threadIdx.x + o];
    __syncthreads();
  }
  if (threadIdx.x == 0) norms[r] = sqrtf(red[0]);
}

__global__ void topk_k(float* __restrict__ norms, int* __restrict__ topidx) {
  __shared__ float sv[1024];
  __shared__ int   si[1024];
  for (int k = 0; k < KPROM; ++k) {
    float bv = -__builtin_inff(); int bi = 0x7fffffff;
    for (int j = threadIdx.x; j < NCAND; j += 1024) {
      float v = norms[j];
      if (v > bv || (v == bv && j < bi)) { bv = v; bi = j; }
    }
    sv[threadIdx.x] = bv; si[threadIdx.x] = bi; __syncthreads();
    for (int o = 512; o > 0; o >>= 1) {
      if (threadIdx.x < o) {
        float v = sv[threadIdx.x + o]; int i2 = si[threadIdx.x + o];
        if (v > sv[threadIdx.x] || (v == sv[threadIdx.x] && i2 < si[threadIdx.x])) {
          sv[threadIdx.x] = v; si[threadIdx.x] = i2;
        }
      }
      __syncthreads();
    }
    if (threadIdx.x == 0) { topidx[k] = si[0]; norms[si[0]] = -__builtin_inff(); }
    __syncthreads();
  }
}

__global__ void s_scatter_k(float* __restrict__ out_s, const float* __restrict__ cand,
                            const int* __restrict__ topidx, const int* __restrict__ sp) {
  int i = blockIdx.x;
  int row = ((*sp) + i) % S_SIZE;
  const float* src = cand + (size_t)topidx[i] * DIM;
  float* dst = out_s + (size_t)row * DIM;
  for (int c = threadIdx.x; c < DIM; c += 256) dst[c] = src[c];
}

// ---------------------------------------------------------------- attention chain
__global__ void lmean_k(const float* __restrict__ l_in, float* __restrict__ lmean) {
  int c = blockIdx.x * 256 + threadIdx.x;
  float s = 0.f;
  for (int r = 0; r < L_SIZE; ++r) s += l_in[(size_t)r * DIM + c];
  lmean[c] = s * (1.0f / (float)L_SIZE);
}

__global__ void qrow_k(const float* __restrict__ wq, const float* __restrict__ bq,
                       const float* __restrict__ lmean, float* __restrict__ qv) {
  __shared__ float red[256];
  int r = blockIdx.x;
  const float* row = wq + (size_t)r * DIM;
  float s = 0.f;
  for (int j = threadIdx.x; j < DIM; j += 256) s += row[j] * lmean[j];
  red[threadIdx.x] = s; __syncthreads();
  for (int o = 128; o > 0; o >>= 1) {
    if (threadIdx.x < o) red[threadIdx.x] += red[threadIdx.x + o];
    __syncthreads();
  }
  if (threadIdx.x == 0) qv[r] = red[0] + bq[r];
}

__global__ void tcol_k(const float* __restrict__ wk, const float* __restrict__ qv,
                       float* __restrict__ tv) {
  int c = blockIdx.x * 256 + threadIdx.x;
  float s = 0.f;
  for (int i = 0; i < DIM; ++i) s += wk[(size_t)i * DIM + c] * qv[i];
  tv[c] = s;
}

// argmax_s( s_mem[s] . t )  — softmax/scale/bias are monotone, argmax-invariant
__global__ void logits_k(const float* __restrict__ out_s, const float* __restrict__ tv,
                         unsigned long long* slots) {
  __shared__ float red[256];
  int s = blockIdx.x;
  const float* row = out_s + (size_t)s * DIM;
  float d = 0.f;
  for (int j = threadIdx.x; j < DIM; j += 256) d += row[j] * tv[j];
  red[threadIdx.x] = d; __syncthreads();
  for (int o = 128; o > 0; o >>= 1) {
    if (threadIdx.x < o) red[threadIdx.x] += red[threadIdx.x + o];
    __syncthreads();
  }
  if (threadIdx.x == 0) atomicMax(&slots[0], packMax(red[0], (unsigned)s));
}

__global__ void cand_prep_k(const float* __restrict__ out_s, const unsigned long long* slots,
                            float* scal, int* iscal, float* __restrict__ cn) {
  __shared__ float red[256];
  __shared__ float snorm;
  unsigned bs = unpackIdx(slots[0]);
  const float* cr = out_s + (size_t)bs * DIM;
  float s = 0.f;
  for (int c = threadIdx.x; c < DIM; c += 256) { float x = cr[c]; s += x * x; }
  red[threadIdx.x] = s; __syncthreads();
  for (int o = 128; o > 0; o >>= 1) {
    if (threadIdx.x < o) red[threadIdx.x] += red[threadIdx.x + o];
    __syncthreads();
  }
  if (threadIdx.x == 0) { snorm = sqrtf(red[0]); scal[2] = snorm; iscal[0] = (int)bs; }
  __syncthreads();
  float dn = fmaxf(snorm, 1e-12f);
  for (int c = threadIdx.x; c < DIM; c += 256) cn[c] = cr[c] / dn;
}

// ---------------------------------------------------------------- M tier prep
__global__ void mmnorm_k(const float* __restrict__ m_in, float* __restrict__ mrn,
                         __bf16* __restrict__ mmbf) {
  __shared__ float red[256];
  __shared__ float snorm;
  int r = blockIdx.x;
  const float* row = m_in + (size_t)r * DIM;
  float s = 0.f;
  for (int c = threadIdx.x; c < DIM; c += 256) { float x = row[c]; s += x * x; }
  red[threadIdx.x] = s; __syncthreads();
  for (int o = 128; o > 0; o >>= 1) {
    if (threadIdx.x < o) red[threadIdx.x] += red[threadIdx.x + o];
    __syncthreads();
  }
  if (threadIdx.x == 0) { snorm = sqrtf(red[0]); mrn[r] = snorm; }
  __syncthreads();
  float dn = fmaxf(snorm, 1e-12f);
  __bf16* dst = mmbf + (size_t)r * DIM;
  for (int c = threadIdx.x; c < DIM; c += 256) dst[c] = (__bf16)(row[c] / dn);
}

__global__ void sims_k(const float* __restrict__ m_in, const float* __restrict__ cn,
                       const float* __restrict__ mrn, unsigned long long* slots) {
  __shared__ float red[256];
  int r = blockIdx.x;
  const float* row = m_in + (size_t)r * DIM;
  float d = 0.f;
  for (int c = threadIdx.x; c < DIM; c += 256) d += row[c] * cn[c];
  red[threadIdx.x] = d; __syncthreads();
  for (int o = 128; o > 0; o >>= 1) {
    if (threadIdx.x < o) red[threadIdx.x] += red[threadIdx.x + o];
    __syncthreads();
  }
  if (threadIdx.x == 0) {
    float sim = red[0] / fmaxf(mrn[r], 1e-12f);
    atomicMax(&slots[1], packMax(sim, (unsigned)r));
  }
}

// ---- fused WMMA GEMM + lower-triangle argmax ----
// One wave per 32x32 macro-tile (2x2 grid of 16x16 WMMA tiles): each A/B
// fragment feeds two v_wmma ops -> 2x arithmetic intensity vs 1 tile/wave.
union BfFrag { uint4 u[2]; v16bf v; };

__global__ __launch_bounds__(32)
void simmat_wmma_k(const __bf16* __restrict__ mmbf, unsigned long long* slot) {
  int bi = blockIdx.x, bj = blockIdx.y;       // 32x32 macro-tile coords
  if (bj > bi) return;                        // lower triangle macro-tiles only
  int lane = threadIdx.x;
  int m16  = lane & 15;                       // A row in tile / B col in tile
  int hg   = lane >> 4;                       // K half-group selector
  const __bf16* a0 = mmbf + (size_t)(bi * 32 + m16) * DIM;        // M rows  0..15
  const __bf16* a1 = a0 + (size_t)16 * DIM;                       // M rows 16..31
  const __bf16* b0 = mmbf + (size_t)(bj * 32 + m16) * DIM;        // N cols  0..15
  const __bf16* b1 = b0 + (size_t)16 * DIM;                       // N cols 16..31
  v8f acc00 = {0.f,0.f,0.f,0.f,0.f,0.f,0.f,0.f};
  v8f acc01 = acc00, acc10 = acc00, acc11 = acc00;
  for (int kb = 0; kb < DIM; kb += 32) {
    BfFrag fa0, fa1, fb0, fb1;
    // 16-bit A-matrix 16x32 layout: lane group hg holds K = kb + hg*8 + {0..7, 16..23}
    const uint4* ap0 = (const uint4*)(a0 + kb + hg * 8);
    const uint4* ap1 = (const uint4*)(a1 + kb + hg * 8);
    fa0.u[0] = ap0[0]; fa0.u[1] = ap0[2];     // +32B -> K offsets +16
    fa1.u[0] = ap1[0]; fa1.u[1] = ap1[2];
    // B-matrix 32x16 layout: lane group hg holds K = kb + hg*8 + {0..7, 16..23}
    const uint4* bp0 = (const uint4*)(b0 + kb + hg * 8);
    const uint4* bp1 = (const uint4*)(b1 + kb + hg * 8);
    fb0.u[0] = bp0[0]; fb0.u[1] = bp0[2];
    fb1.u[0] = bp1[0]; fb1.u[1] = bp1[2];
    if (kb + 32 < DIM) {
      __builtin_prefetch(a0 + kb + 32 + hg * 8, 0, 3);
      __builtin_prefetch(a1 + kb + 32 + hg * 8, 0, 3);
      __builtin_prefetch(b0 + kb + 32 + hg * 16, 0, 3);
      __builtin_prefetch(b1 + kb + 32 + hg * 16, 0, 3);
    }
    acc00 = __builtin_amdgcn_wmma_f32_16x16x32_bf16(false, fa0.v, false, fb0.v,
                                                    (short)0, acc00, false, false);
    acc01 = __builtin_amdgcn_wmma_f32_16x16x32_bf16(false, fa0.v, false, fb1.v,
                                                    (short)0, acc01, false, false);
    acc10 = __builtin_amdgcn_wmma_f32_16x16x32_bf16(false, fa1.v, false, fb0.v,
                                                    (short)0, acc10, false, false);
    acc11 = __builtin_amdgcn_wmma_f32_16x16x32_bf16(false, fa1.v, false, fb1.v,
                                                    (short)0, acc11, false, false);
  }
  // C/D layout: lane holds N = lane&15, M = v + 8*(lane>>4) for VGPR v
  unsigned long long best = 0ull;
#pragma unroll
  for (int mt = 0; mt < 2; ++mt) {
#pragma unroll
    for (int nt = 0; nt < 2; ++nt) {
      v8f acc = (mt == 0) ? (nt == 0 ? acc00 : acc01)
                          : (nt == 0 ? acc10 : acc11);
#pragma unroll
      for (int v = 0; v < 8; ++v) {
        int gm = bi * 32 + mt * 16 + v + 8 * hg;
        int gn = bj * 32 + nt * 16 + m16;
        float val = (gm > gn) ? acc[v] : -1.0f;  // triu (incl. diag) masked to -1
        unsigned long long k = packMax(val, (unsigned)(gm * M_SIZE + gn));
        if (k > best) best = k;
      }
    }
  }
  for (int off = 16; off > 0; off >>= 1) {
    unsigned long long o = __shfl_xor(best, off, 32);
    if (o > best) best = o;
  }
  if (lane == 0) atomicMax(slot, best);
}

// ---------------------------------------------------------------- reductions
__global__ void mean_k(const float* __restrict__ v, int n, float* dst) {
  __shared__ float red[256];
  float s = 0.f;
  for (int i = threadIdx.x; i < n; i += 256) s += v[i];
  red[threadIdx.x] = s; __syncthreads();
  for (int o = 128; o > 0; o >>= 1) {
    if (threadIdx.x < o) red[threadIdx.x] += red[threadIdx.x + o];
    __syncthreads();
  }
  if (threadIdx.x == 0) dst[0] = red[0] / (float)n;
}

__global__ void argext_k(const float* __restrict__ v, int n, int negate,
                         unsigned long long* slot) {
  __shared__ unsigned long long red[256];
  int i = blockIdx.x * 256 + threadIdx.x;
  unsigned long long k = 0ull;
  if (i < n) {
    float x = negate ? -v[i] : v[i];
    k = packMax(x, (unsigned)i);
  }
  red[threadIdx.x] = k; __syncthreads();
  for (int o = 128; o > 0; o >>= 1) {
    if (threadIdx.x < o && red[threadIdx.x + o] > red[threadIdx.x])
      red[threadIdx.x] = red[threadIdx.x + o];
    __syncthreads();
  }
  if (threadIdx.x == 0) atomicMax(slot, red[0]);
}

// ---------------------------------------------------------------- M-tier branch
__global__ void decision_k(const float* __restrict__ m_in, const float* __restrict__ mu_in,
                           float* __restrict__ out_m, float* __restrict__ out_mu,
                           const float* __restrict__ out_s, const float* __restrict__ mrn,
                           const unsigned long long* slots, const float* scal,
                           const int* iscal) {
  __shared__ float red[1024];
  __shared__ float shn;
  int t = threadIdx.x;
  float    sim_cand = unpackVal(slots[1]);
  unsigned msi      = unpackIdx(slots[1]);
  float    internal = unpackVal(slots[2]);
  unsigned flat     = unpackIdx(slots[2]);
  unsigned idx1 = flat / M_SIZE, idx2 = flat % M_SIZE;
  float mu_mean = scal[0];
  float cnrm    = scal[2];
  const float* cand = out_s + (size_t)iscal[0] * DIM;

  if (sim_cand > 0.98f) {                     // merge candidate into msi
    float tmp = (m_in[(size_t)msi * DIM + t] + cand[t]) * 0.5f;
    red[t] = tmp * tmp; __syncthreads();
    for (int o = 512; o > 0; o >>= 1) { if (t < o) red[t] += red[t + o]; __syncthreads(); }
    if (t == 0) shn = fmaxf(sqrtf(red[0]), 1e-12f);
    __syncthreads();
    out_m[(size_t)msi * DIM + t] = tmp / shn;
    if (t == 0) out_mu[msi] = (mu_in[msi] + mu_mean) * 0.5f;
  } else if (internal > 0.98f) {              // internal pair merge
    float tmp = (m_in[(size_t)idx1 * DIM + t] + m_in[(size_t)idx2 * DIM + t]) * 0.5f;
    red[t] = tmp * tmp; __syncthreads();
    for (int o = 512; o > 0; o >>= 1) { if (t < o) red[t] += red[t + o]; __syncthreads(); }
    if (t == 0) shn = fmaxf(sqrtf(red[0]), 1e-12f);
    __syncthreads();
    out_m[(size_t)idx1 * DIM + t] = tmp / shn;
    out_m[(size_t)idx2 * DIM + t] = cand[t];
    if (t == 0) {
      float nm1 = (mu_in[idx1] + mu_in[idx2]) * 0.5f;
      out_mu[idx1] = nm1;
      float mean_after = mu_mean + (nm1 - mu_in[idx1]) / (float)M_SIZE;
      out_mu[idx2] = mean_after + 1e-5f;
    }
  } else {                                    // replace weakest if stronger
    unsigned li = unpackIdx(slots[3]);
    if (cnrm > mrn[li]) {
      out_m[(size_t)li * DIM + t] = cand[t];
      if (t == 0) out_mu[li] = mu_mean + 1e-5f;
    }
  }
}

// ---------------------------------------------------------------- L tier
__global__ void ltier_k(const float* __restrict__ l_in, const float* __restrict__ out_m,
                        float* __restrict__ out_l, float* __restrict__ out_lu,
                        float* __restrict__ out_sp, const unsigned long long* slots,
                        const float* scal, const int* sp) {
  int t = threadIdx.x;
  unsigned mi = unpackIdx(slots[5]);          // argmax of post-update m_utility
  unsigned lj = unpackIdx(slots[4]);          // argmin of original l_utility
  out_l[(size_t)lj * DIM + t] =
      0.9f * l_in[(size_t)lj * DIM + t] + 0.1f * out_m[(size_t)mi * DIM + t];
  if (t == 0) {
    out_lu[lj] = scal[1];                     // mean of pre-write l_utility
    out_sp[0] = (float)(((*sp) + KPROM) % S_SIZE);
  }
}

// ================================================================ host launch
extern "C" void kernel_launch(void* const* d_in, const int* in_sizes, int n_in,
                              void* d_out, int out_size, void* d_ws, size_t ws_size,
                              hipStream_t stream) {
  const float* cand  = (const float*)d_in[0];
  const float* s_in  = (const float*)d_in[1];
  const float* m_in  = (const float*)d_in[2];
  const float* l_in  = (const float*)d_in[3];
  const float* mu_in = (const float*)d_in[4];
  const float* lu_in = (const float*)d_in[5];
  const float* wq    = (const float*)d_in[6];
  const float* bq    = (const float*)d_in[7];
  const float* wk    = (const float*)d_in[8];
  const float* bk    = (const float*)d_in[9];
  const int*   sp    = (const int*)  d_in[10];
  (void)bk; (void)in_sizes; (void)n_in; (void)out_size; (void)ws_size;

  float* out    = (float*)d_out;
  float* out_s  = out;
  float* out_m  = out_s + (size_t)S_SIZE * DIM;
  float* out_l  = out_m + (size_t)M_SIZE * DIM;
  float* out_mu = out_l + (size_t)L_SIZE * DIM;
  float* out_lu = out_mu + M_SIZE;
  float* out_sp = out_lu + L_SIZE;

  char* ws = (char*)d_ws;
  unsigned long long* slots = (unsigned long long*)(ws + WS_SLOTS);
  float*  scal   = (float*)(ws + WS_SCAL);
  int*    iscal  = (int*)  (ws + WS_ISCAL);
  float*  norms  = (float*)(ws + WS_NORMS);
  int*    topidx = (int*)  (ws + WS_TOPIDX);
  float*  lmean  = (float*)(ws + WS_LMEAN);
  float*  qv     = (float*)(ws + WS_Q);
  float*  tv     = (float*)(ws + WS_T);
  float*  cn     = (float*)(ws + WS_CN);
  float*  mrn    = (float*)(ws + WS_MRN);
  __bf16* mmbf   = (__bf16*)(ws + WS_MBF);

  zero_slots_k<<<1, 32, 0, stream>>>(slots);

  auto launch_copy = [&](float* dst, const float* src, size_t n) {
    size_t n4 = n / 4;
    int blocks = (int)((n4 + 255) / 256);
    if (blocks > 8192) blocks = 8192;
    copy4_k<<<blocks, 256, 0, stream>>>((float4*)dst, (const float4*)src, n4);
  };
  launch_copy(out_s,  s_in,  (size_t)S_SIZE * DIM);
  launch_copy(out_m,  m_in,  (size_t)M_SIZE * DIM);
  launch_copy(out_l,  l_in,  (size_t)L_SIZE * DIM);
  launch_copy(out_mu, mu_in, M_SIZE);
  launch_copy(out_lu, lu_in, L_SIZE);

  cand_norms_k<<<NCAND, 256, 0, stream>>>(cand, norms);
  topk_k<<<1, 1024, 0, stream>>>(norms, topidx);
  s_scatter_k<<<KPROM, 256, 0, stream>>>(out_s, cand, topidx, sp);

  lmean_k<<<DIM / 256, 256, 0, stream>>>(l_in, lmean);
  qrow_k<<<DIM, 256, 0, stream>>>(wq, bq, lmean, qv);
  tcol_k<<<DIM / 256, 256, 0, stream>>>(wk, qv, tv);
  logits_k<<<S_SIZE, 256, 0, stream>>>(out_s, tv, slots);
  cand_prep_k<<<1, 256, 0, stream>>>(out_s, slots, scal, iscal, cn);

  mmnorm_k<<<M_SIZE, 256, 0, stream>>>(m_in, mrn, mmbf);
  sims_k<<<M_SIZE, 256, 0, stream>>>(m_in, cn, mrn, slots);

  dim3 g(M_SIZE / 32, M_SIZE / 32);           // 32x32 macro-tiles
  simmat_wmma_k<<<g, 32, 0, stream>>>(mmbf, slots + 2);

  mean_k<<<1, 256, 0, stream>>>(mu_in, M_SIZE, scal + 0);
  mean_k<<<1, 256, 0, stream>>>(lu_in, L_SIZE, scal + 1);
  argext_k<<<(M_SIZE + 255) / 256, 256, 0, stream>>>(mu_in, M_SIZE, 1, slots + 3);
  argext_k<<<(L_SIZE + 255) / 256, 256, 0, stream>>>(lu_in, L_SIZE, 1, slots + 4);

  decision_k<<<1, 1024, 0, stream>>>(m_in, mu_in, out_m, out_mu, out_s, mrn,
                                     slots, scal, iscal);
  argext_k<<<(M_SIZE + 255) / 256, 256, 0, stream>>>(out_mu, M_SIZE, 0, slots + 5);
  ltier_k<<<1, 1024, 0, stream>>>(l_in, out_m, out_l, out_lu, out_sp, slots, scal, sp);
}
